// MyGRU_77653008712320
// MI455X (gfx1250) — matrix operand order repaired and checked by reading, not compiled, not run on hardware
//
#include <hip/hip_runtime.h>
#include <hip/hip_bf16.h>

#define S_LEN 1024
#define B_DIM 64
#define I_DIM 512
#define H_DIM 512
#define NWG   16   // persistent scan workgroups == max cluster size

typedef __attribute__((ext_vector_type(16))) __bf16 bf16x16;
typedef __attribute__((ext_vector_type(8)))  float  f32x8;

// TDM availability: device pass on gfx1250 toolchains declares the builtin;
// host pass (x86) reports 0 and takes the manual-copy path (parse-safe).
#ifndef HAVE_TDM
#  if defined(__has_builtin)
#    if __has_builtin(__builtin_amdgcn_tensor_load_to_lds)
#      define HAVE_TDM 1
#    else
#      define HAVE_TDM 0
#    endif
#  else
#    define HAVE_TDM 0
#  endif
#endif

#if HAVE_TDM
typedef __attribute__((ext_vector_type(4))) unsigned int tdm_u32x4;
typedef __attribute__((ext_vector_type(4))) int          tdm_i32x4;
typedef __attribute__((ext_vector_type(8))) int          tdm_i32x8;
#endif

// ---------- helpers ----------

__device__ __forceinline__ unsigned short f2bf(float f) {
  unsigned u = __float_as_uint(f);
  u += 0x7FFFu + ((u >> 16) & 1u);        // round-to-nearest-even
  return (unsigned short)(u >> 16);
}

// A-fragment (16x32 bf16, row-major source):
// lane 0-15: K = 0..15; lane 16-31: K = 16..31.
// caller passes p already offset by row*ld + kbase + ((lane>>4)<<4)
__device__ __forceinline__ bf16x16 frag_a(const unsigned short* p) {
  bf16x16 a;
  uint4* v = reinterpret_cast<uint4*>(&a);
  v[0] = *reinterpret_cast<const uint4*>(p);        // K+0..7
  v[1] = *reinterpret_cast<const uint4*>(p + 8);    // K+8..15
  return a;
}

// B-fragment (32x16 bf16). B[k][n] = U[n][k] (row-major U) so each lane reads
// 16 contiguous bf16 of row n. lane 0-15: K=0..15, lane 16-31: K=16..31.
// caller passes p = row_n_base + kbase + ((lane>>4)<<4)
__device__ __forceinline__ bf16x16 frag_b(const unsigned short* p) {
  bf16x16 b;
  uint4* v = reinterpret_cast<uint4*>(&b);
  v[0] = *reinterpret_cast<const uint4*>(p);
  v[1] = *reinterpret_cast<const uint4*>(p + 8);
  return b;
}

// ---------- fp32 -> bf16 conversion ----------

__global__ void cvt_bf16(const float* __restrict__ in, unsigned short* __restrict__ out, long n) {
  long i = (long)blockIdx.x * blockDim.x + threadIdx.x;
  long stride = (long)gridDim.x * blockDim.x;
  for (; i < n; i += stride) out[i] = f2bf(in[i]);
}

// ---------- init: h state + barrier counters ----------

__global__ void gru_init(const float* __restrict__ h0, unsigned short* __restrict__ hbf,
                         float* __restrict__ hf32, unsigned* __restrict__ bar) {
  int i = blockIdx.x * 256 + threadIdx.x;
  if (i == 0) { bar[0] = 0u; bar[1] = 0u; }
  if (i < B_DIM * H_DIM) {
    float v = h0[i];
    hf32[i] = v;
    hbf[i]  = f2bf(v);
  }
}

// ---------- input projection GEMM: [65536 x 512] * [512 x 1536] ----------

__global__ __launch_bounds__(256) void gru_xgemm(
    const unsigned short* __restrict__ xbf,  // S*B x I, bf16
    const unsigned short* __restrict__ Wbf,  // 3H x I, bf16 (Wz|Wr|Wh rows)
    const float* __restrict__ bz, const float* __restrict__ br, const float* __restrict__ bh,
    float* __restrict__ xz, float* __restrict__ xr, float* __restrict__ xh) {
  const int tid  = threadIdx.x;
  const int wave = tid >> 5;
  const int lane = tid & 31;

  const int mbase = blockIdx.x * 64 + (wave & 3) * 16;      // 4 M-tiles per WG
  const int nbase = blockIdx.y * 128 + (wave >> 2) * 64;    // each wave: 4 N-tiles
  const int a_koff = (lane >> 4) << 4;
  const int b_koff = (lane >> 4) << 4;

  const unsigned short* arow = xbf + (size_t)(mbase + (lane & 15)) * I_DIM + a_koff;
  const unsigned short* b0 = Wbf + (size_t)(nbase +  0 + (lane & 15)) * I_DIM + b_koff;
  const unsigned short* b1 = Wbf + (size_t)(nbase + 16 + (lane & 15)) * I_DIM + b_koff;
  const unsigned short* b2 = Wbf + (size_t)(nbase + 32 + (lane & 15)) * I_DIM + b_koff;
  const unsigned short* b3 = Wbf + (size_t)(nbase + 48 + (lane & 15)) * I_DIM + b_koff;

  f32x8 acc0 = (f32x8)0.0f, acc1 = (f32x8)0.0f, acc2 = (f32x8)0.0f, acc3 = (f32x8)0.0f;

#pragma unroll 4
  for (int kt = 0; kt < 16; ++kt) {
    const int k = kt * 32;
    bf16x16 a = frag_a(arow + k);
    __builtin_prefetch(arow + k + 128, 0, 3);   // stream-ahead into near cache
    acc0 = __builtin_amdgcn_wmma_f32_16x16x32_bf16(false, a, false, frag_b(b0 + k), (short)0, acc0, false, false);
    acc1 = __builtin_amdgcn_wmma_f32_16x16x32_bf16(false, a, false, frag_b(b1 + k), (short)0, acc1, false, false);
    acc2 = __builtin_amdgcn_wmma_f32_16x16x32_bf16(false, a, false, frag_b(b2 + k), (short)0, acc2, false, false);
    acc3 = __builtin_amdgcn_wmma_f32_16x16x32_bf16(false, a, false, frag_b(b3 + k), (short)0, acc3, false, false);
  }

  const int gate = (blockIdx.y * 128) / H_DIM;              // block lies in one gate
  float* outp        = gate == 0 ? xz : (gate == 1 ? xr : xh);
  const float* bias  = gate == 0 ? bz : (gate == 1 ? br : bh);
  const int row0 = mbase + ((lane >> 4) << 3);

  auto epi = [&](f32x8& acc, int jn) {
    const int c = nbase + jn * 16 + (lane & 15) - gate * H_DIM;
    const float bv = bias[c];
#pragma unroll
    for (int r = 0; r < 8; ++r)
      outp[(size_t)(row0 + r) * H_DIM + c] = acc[r] + bv;
  };
  epi(acc0, 0); epi(acc1, 1); epi(acc2, 2); epi(acc3, 3);
}

// ---------- device-wide barrier for the persistent scan ----------
// Fast path: CDNA5 cluster barrier (s_barrier_signal/-wait -3) when the 16 WGs
// are launched as one cluster; atomic generation barrier guarantees correctness
// under a plain launch (cluster ops are NOP outside a cluster).
__device__ __forceinline__ void grid_barrier(unsigned* bar) {
  __syncthreads();
  __threadfence();                                   // publish this WG's writes
  if (threadIdx.x == 0) {
    unsigned gen = __hip_atomic_load(&bar[1], __ATOMIC_RELAXED, __HIP_MEMORY_SCOPE_AGENT);
    if (__hip_atomic_fetch_add(&bar[0], 1u, __ATOMIC_ACQ_REL, __HIP_MEMORY_SCOPE_AGENT) == NWG - 1) {
      __hip_atomic_store(&bar[0], 0u, __ATOMIC_RELAXED, __HIP_MEMORY_SCOPE_AGENT);
      __hip_atomic_fetch_add(&bar[1], 1u, __ATOMIC_RELEASE, __HIP_MEMORY_SCOPE_AGENT);
    } else {
      while (__hip_atomic_load(&bar[1], __ATOMIC_ACQUIRE, __HIP_MEMORY_SCOPE_AGENT) == gen)
        __builtin_amdgcn_s_sleep(2);
    }
  }
  if (threadIdx.x == 0) __builtin_amdgcn_s_cluster_barrier();
  __syncthreads();
  __threadfence();                                   // acquire peers' writes
}

// ---------- persistent recurrent scan: 16 WGs x 8 waves ----------

__global__ __launch_bounds__(256) void gru_scan(
    const unsigned short* __restrict__ Ubf,  // 3H x H bf16 (Uz|Ur|Uh rows)
    const float* __restrict__ ubz, const float* __restrict__ ubr, const float* __restrict__ ubh,
    const float* __restrict__ xz, const float* __restrict__ xr, const float* __restrict__ xh,
    unsigned short* __restrict__ hbf, unsigned short* __restrict__ rhbf,
    float* __restrict__ hf32, float* __restrict__ out, unsigned* __restrict__ bar) {
  // 64KB LDS: Uz and Ur slices (32 rows x 512 K each), reused for all 1024 steps.
  __shared__ __align__(16) unsigned short ldsU[2 * 32 * H_DIM];

  const int wg   = blockIdx.x;             // owns columns [wg*32, wg*32+32)
  const int tid  = threadIdx.x;
  const int wave = tid >> 5;
  const int lane = tid & 31;

#if HAVE_TDM
  // Stage Uz/Ur slice via the Tensor Data Mover: one 3D-tile DMA per WG.
  // Tile: dim0 = 512 K (contiguous bf16), dim1 = 32 rows (stride 512 elems),
  //       dim2 = 2 gates (stride 512*512 elems). data_size = 2 bytes.
  if (tid < 32) {                          // one wave issues; EXEC is ignored by TDM
    const unsigned long long ga =
        (unsigned long long)(const void*)(Ubf + (size_t)wg * 32 * H_DIM);
    const unsigned ldsa = (unsigned)(unsigned long long)(const void*)&ldsU[0];
    tdm_u32x4 g0;
    g0[0] = 1u;                                            // count=1, user descriptor
    g0[1] = ldsa;                                          // lds_addr
    g0[2] = (unsigned)ga;                                  // global_addr[31:0]
    g0[3] = (unsigned)((ga >> 32) & 0x01FFFFFFu) | (2u << 30);  // ga[56:32] | type=2
    tdm_i32x8 g1;
    g1[0] = 0x00010000;                    // data_size=2B, no multicast/barrier/pad
    g1[1] = (int)(H_DIM << 16);            // tensor_dim0[15:0] at bits[63:48]
    g1[2] = (int)((H_DIM >> 16) | ((unsigned)H_DIM << 16)); // dim0 hi | tensor_dim1 lo
    g1[3] = (int)((H_DIM >> 16) | ((unsigned)H_DIM << 16)); // dim1 hi | tile_dim0=512
    g1[4] = (int)(32u | (2u << 16));       // tile_dim1=32 rows, tile_dim2=2 gates
    g1[5] = (int)H_DIM;                    // tensor_dim0_stride[31:0] = 512
    g1[6] = (int)(((unsigned)(H_DIM * H_DIM) & 0xFFFFu) << 16); // stride0 hi | stride1 lo
    g1[7] = (int)((unsigned)(H_DIM * H_DIM) >> 16);             // stride1[47:16]
    tdm_i32x4 g2;
    g2[0] = 3;                             // tensor_dim2 (3 gates in memory)
    g2[1] = 0; g2[2] = 0; g2[3] = 0;       // no dim3, no iterate
    tdm_i32x4 g3 = (tdm_i32x4)0;
#if __clang_major__ >= 23
    tdm_i32x8 g4 = (tdm_i32x8)0;
    __builtin_amdgcn_tensor_load_to_lds(g0, g1, g2, g3, g4, 0);
#else
    __builtin_amdgcn_tensor_load_to_lds(g0, g1, g2, g3, 0);
#endif
    __builtin_amdgcn_s_wait_tensorcnt(0);
  }
#else
  {
    const uint4* Uu4 = reinterpret_cast<const uint4*>(Ubf);
    uint4* Lu4 = reinterpret_cast<uint4*>(ldsU);
    for (int c = tid; c < 2 * 32 * (H_DIM / 8); c += 256) {   // 64 uint4 per row
      const int r = c >> 6, pos = c & 63;
      const int gate = r >> 5, nloc = r & 31;
      Lu4[c] = Uu4[(size_t)(gate * H_DIM + wg * 32 + nloc) * (H_DIM / 8) + pos];
    }
  }
#endif
  __syncthreads();

  const int m  = wave & 3;                 // batch tile (64 rows -> 4 tiles)
  const int nt = wave >> 2;                // 0/1: which 16-col tile of the slice
  const int row_lo = m * 16 + ((lane >> 4) << 3);
  const int colg   = wg * 32 + nt * 16 + (lane & 15);
  const int a_koff = (lane >> 4) << 4;
  const int b_koff = (lane >> 4) << 4;

  const unsigned short* aH  = hbf  + (size_t)(m * 16 + (lane & 15)) * H_DIM + a_koff;
  const unsigned short* aRH = rhbf + (size_t)(m * 16 + (lane & 15)) * H_DIM + a_koff;
  const unsigned short* ldsUz = &ldsU[(size_t)(0 * 32 + nt * 16 + (lane & 15)) * H_DIM] + b_koff;
  const unsigned short* ldsUr = &ldsU[(size_t)(1 * 32 + nt * 16 + (lane & 15)) * H_DIM] + b_koff;
  const unsigned short* Uh_row = Ubf + (size_t)(2 * H_DIM + colg) * H_DIM + b_koff;  // L2-resident

  const float biasZ = ubz[colg], biasR = ubr[colg], biasH = ubh[colg];

  for (int s = 0; s < S_LEN; ++s) {
    const size_t xoff = (size_t)s * B_DIM * H_DIM;

    // ---- phase 1: z and r gates over K = H ----
    f32x8 accz = (f32x8)0.0f, accr = (f32x8)0.0f;
#pragma unroll 4
    for (int kt = 0; kt < 16; ++kt) {
      const int k = kt * 32;
      bf16x16 a = frag_a(aH + k);
      accz = __builtin_amdgcn_wmma_f32_16x16x32_bf16(false, a, false, frag_b(ldsUz + k), (short)0, accz, false, false);
      accr = __builtin_amdgcn_wmma_f32_16x16x32_bf16(false, a, false, frag_b(ldsUr + k), (short)0, accr, false, false);
    }

    float zv[8], hold[8];
#pragma unroll
    for (int r = 0; r < 8; ++r) {
      const size_t idx = (size_t)(row_lo + r) * H_DIM + colg;
      float z = accz[r] + xz[xoff + idx] + biasZ;
      z = 1.0f / (1.0f + __expf(-z));
      float rr = accr[r] + xr[xoff + idx] + biasR;
      rr = 1.0f / (1.0f + __expf(-rr));
      const float hv = hf32[idx];
      zv[r] = z; hold[r] = hv;
      rhbf[idx] = f2bf(rr * hv);           // broadcast r ⊙ h for phase 2
    }

    grid_barrier(bar);                     // all r⊙h visible device-wide

    // ---- phase 2: candidate gate over K = H ----
    f32x8 acch = (f32x8)0.0f;
#pragma unroll 4
    for (int kt = 0; kt < 16; ++kt) {
      const int k = kt * 32;
      bf16x16 a = frag_a(aRH + k);
      acch = __builtin_amdgcn_wmma_f32_16x16x32_bf16(false, a, false, frag_b(Uh_row + k), (short)0, acch, false, false);
    }

#pragma unroll
    for (int r = 0; r < 8; ++r) {
      const size_t idx = (size_t)(row_lo + r) * H_DIM + colg;
      const float pre = acch[r] + xh[xoff + idx] + biasH;
      const float e = __expf(2.0f * pre);
      const float ht = (e - 1.0f) / (e + 1.0f);          // tanh
      const float hn = hold[r] + zv[r] * (ht - hold[r]); // (1-z)h + z*h~
      out[xoff + idx] = hn;
      hf32[idx] = hn;
      hbf[idx]  = f2bf(hn);
      if (s == S_LEN - 1)
        out[(size_t)S_LEN * B_DIM * H_DIM + idx] = hn;   // hx
    }

    grid_barrier(bar);                     // h_new visible before next step
  }
}

// ---------- host launch ----------

extern "C" void kernel_launch(void* const* d_in, const int* in_sizes, int n_in,
                              void* d_out, int out_size, void* d_ws, size_t ws_size,
                              hipStream_t stream) {
  (void)in_sizes; (void)n_in; (void)out_size; (void)ws_size;

  const float* x    = (const float*)d_in[0];
  const float* h0   = (const float*)d_in[1];
  const float* Wz_w = (const float*)d_in[2];  const float* Wz_b = (const float*)d_in[3];
  const float* Uz_w = (const float*)d_in[4];  const float* Uz_b = (const float*)d_in[5];
  const float* Wr_w = (const float*)d_in[6];  const float* Wr_b = (const float*)d_in[7];
  const float* Ur_w = (const float*)d_in[8];  const float* Ur_b = (const float*)d_in[9];
  const float* Wh_w = (const float*)d_in[10]; const float* Wh_b = (const float*)d_in[11];
  const float* Uh_w = (const float*)d_in[12]; const float* Uh_b = (const float*)d_in[13];

  char* ws = (char*)d_ws;
  size_t off = 0;
  auto carve = [&](size_t bytes) -> void* {
    void* p = ws + off;
    off = (off + bytes + 255) & ~(size_t)255;
    return p;
  };

  unsigned short* xbf  = (unsigned short*)carve((size_t)S_LEN * B_DIM * I_DIM * 2);
  unsigned short* Wbf  = (unsigned short*)carve((size_t)3 * H_DIM * I_DIM * 2);
  unsigned short* Ubf  = (unsigned short*)carve((size_t)3 * H_DIM * H_DIM * 2);
  float* xz            = (float*)carve((size_t)S_LEN * B_DIM * H_DIM * 4);
  float* xr            = (float*)carve((size_t)S_LEN * B_DIM * H_DIM * 4);
  float* xh            = (float*)carve((size_t)S_LEN * B_DIM * H_DIM * 4);
  unsigned short* hbf  = (unsigned short*)carve((size_t)B_DIM * H_DIM * 2);
  unsigned short* rhbf = (unsigned short*)carve((size_t)B_DIM * H_DIM * 2);
  float* hf32          = (float*)carve((size_t)B_DIM * H_DIM * 4);
  unsigned* bar        = (unsigned*)carve(256);

  const long nx = (long)S_LEN * B_DIM * I_DIM;
  const long nw = (long)H_DIM * I_DIM;
  cvt_bf16<<<4096, 256, 0, stream>>>(x, xbf, nx);
  cvt_bf16<<<512, 256, 0, stream>>>(Wz_w, Wbf + 0 * nw, nw);
  cvt_bf16<<<512, 256, 0, stream>>>(Wr_w, Wbf + 1 * nw, nw);
  cvt_bf16<<<512, 256, 0, stream>>>(Wh_w, Wbf + 2 * nw, nw);
  cvt_bf16<<<512, 256, 0, stream>>>(Uz_w, Ubf + 0 * nw, nw);
  cvt_bf16<<<512, 256, 0, stream>>>(Ur_w, Ubf + 1 * nw, nw);
  cvt_bf16<<<512, 256, 0, stream>>>(Uh_w, Ubf + 2 * nw, nw);

  gru_init<<<(B_DIM * H_DIM + 255) / 256, 256, 0, stream>>>(h0, hbf, hf32, bar);

  dim3 gg(S_LEN * B_DIM / 64, (3 * H_DIM) / 128);
  gru_xgemm<<<gg, 256, 0, stream>>>(xbf, Wbf, Wz_b, Wr_b, Wh_b, xz, xr, xh);

  gru_scan<<<NWG, 256, 0, stream>>>(Ubf, Uz_b, Ur_b, Uh_b, xz, xr, xh,
                                    hbf, rhbf, hf32, (float*)d_out, bar);
}
